// GarmentDiffusionMamba_36945308680374
// MI455X (gfx1250) — matrix operand order repaired and checked by reading, not compiled, not run on hardware
//
#include <hip/hip_runtime.h>
#include <hip/hip_bf16.h>
#include <math.h>

// ---------------- model constants ----------------
#define D_MODEL   768
#define D_STATE   16
#define D_CONV    4
#define D_INNER   1536            // 2*D_MODEL
#define DT_RANK   48
#define DT_PAD    64              // DT_RANK padded to mult of 32 for WMMA K
#define N_LAYERS  2
#define N_HEADS   8
#define DH        96
#define SEQL      1443            // 37*39
#define COND_LEN  256
#define BATCH     2
#define EDGE_DIM  9
#define MROWS     (BATCH*SEQL)    // 2886
#define MP        2896            // MROWS padded to multiple of 16
#define MC        (BATCH*COND_LEN)// 512
#define XPROJ_N   80              // dt_rank + 2*d_state
#define XPROJ_NP  128             // padded to multiple of 64 for branch-free GEMM
#define DBC_LD    128

#define CDIV(a,b) (((a)+(b)-1)/(b))

typedef __attribute__((ext_vector_type(16))) _Float16 v16h;
typedef __attribute__((ext_vector_type(8)))  float    v8f;

// ---------------- device helpers ----------------
__device__ __forceinline__ float siluf(float x) { return x / (1.0f + __expf(-x)); }

// =================================================================
// WMMA GEMM:  C[M,N] = A[M,K] * W[N,K]^T  (+ epilogue)
//   A : f16 row-major, lda = K ;  W : f16 row-major [N,K], ldw = K
// Block = 256 threads = 8 waves; block tile 32(M) x 256(N);
// each wave: 16x64 -> 1 A-frag reused over 4 B-frags -> 4 v_wmma / k-step.
// Requirements: K % 64 == 0, N % 64 == 0, M rows padded to 16.
// Wave-uniform tile indices via readfirstlane => pure scalar control flow,
// no EXEC predication around WMMA. K-loop is ping-pong double-buffered so
// next tile's global_load_b128s overlap current tile's matrix ops.
// EPI: 0 f32 | 1 f32+bias | 2 f16 bias+GELU | 3 f32 bias+residual
//      4 f32 bias+softplus | 5 f16 | 6 f16+bias
// =================================================================
template<int EPI>
__global__ __launch_bounds__(256)
void gemm_wmma_f16(const _Float16* __restrict__ A, int lda,
                   const _Float16* __restrict__ W, int ldw,
                   int M, int N, int K,
                   const float* __restrict__ bias,
                   const float* R, int ldr,
                   float* Cf, int ldc,
                   _Float16* Ch, int ldch)
{
    const int lane = threadIdx.x & 31;
    const int wave = __builtin_amdgcn_readfirstlane((int)threadIdx.x) >> 5; // SGPR
    const int m0 = blockIdx.y * 32  + (wave >> 2) * 16;
    const int n0 = blockIdx.x * 256 + (wave & 3) * 64;
    if (m0 >= M || n0 >= N) return;            // scalar branch

    const int hi = lane >> 4;                  // half-wave (K sub-slice select)
    const int lr = lane & 15;                  // row (A) / col (B,C) within tile

    const _Float16* Ar = A + (size_t)(m0 + lr) * lda;
    const _Float16* W0 = W + (size_t)(n0 + lr) * ldw;

    v8f acc[4];
#pragma unroll
    for (int j = 0; j < 4; ++j) acc[j] = v8f{0.f,0.f,0.f,0.f,0.f,0.f,0.f,0.f};

    union F { uint4 u[2]; v16h v; };
    F a0, a1, b0[4], b1[4];

    // A frag 16x32: half-wave 0 holds K k..k+7 & k+16..k+23; hw 1: +8 / +24
    // B frag 32x16: half-wave 0 holds K k..k+7 & k+16..k+23; hw 1: +8 / +24
#define LOAD_A(dst, kk) do { const int ka_ = (kk) + hi * 8;                       \
        dst.u[0] = *(const uint4*)(Ar + ka_);                                     \
        dst.u[1] = *(const uint4*)(Ar + ka_ + 16); } while (0)
#define LOAD_B(dst, kk) do { const int kb_ = (kk) + hi * 8;                       \
        _Pragma("unroll")                                                         \
        for (int j = 0; j < 4; ++j) {                                             \
            const _Float16* Wr_ = W0 + (size_t)(16 * j) * ldw + kb_;              \
            dst[j].u[0] = *(const uint4*)(Wr_);                                   \
            dst[j].u[1] = *(const uint4*)(Wr_ + 16); } } while (0)
#define MMA(af, bf) do { _Pragma("unroll")                                        \
        for (int j = 0; j < 4; ++j)                                               \
            acc[j] = __builtin_amdgcn_wmma_f32_16x16x32_f16(                      \
                false, af.v, false, bf[j].v, (short)0, acc[j], false, false);     \
        } while (0)

    LOAD_A(a0, 0); LOAD_B(b0, 0);
    for (int k = 0; k < K; k += 64) {          // K % 64 == 0 guaranteed
        LOAD_A(a1, k + 32); LOAD_B(b1, k + 32);
        MMA(a0, b0);
        if (k + 64 < K) { LOAD_A(a0, k + 64); LOAD_B(b0, k + 64); }
        MMA(a1, b1);
    }
#undef LOAD_A
#undef LOAD_B
#undef MMA

    // ---- epilogue: lane lr holds col n0+16j+lr; VGPR r holds row m0 + 8*hi + r
#pragma unroll
    for (int j = 0; j < 4; ++j) {
        const int col = n0 + j * 16 + lr;
        float bv = 0.0f;
        if (EPI == 1 || EPI == 2 || EPI == 3 || EPI == 4 || EPI == 6) bv = bias[col];
#pragma unroll
        for (int r = 0; r < 8; ++r) {
            const int row = m0 + hi * 8 + r;
            float v = acc[j][r] + bv;
            if (EPI == 2) v = 0.5f * v * (1.0f + erff(v * 0.70710678118654752f));
            if (EPI == 4) v = (v > 20.0f) ? v : log1pf(__expf(v));
            if (EPI == 3) v += R[(size_t)row * ldr + col];
            if (EPI == 0 || EPI == 1 || EPI == 3 || EPI == 4)
                Cf[(size_t)row * ldc + col] = v;
            else
                Ch[(size_t)row * ldch + col] = (_Float16)v;
        }
    }
}

// ---------------- small elementwise kernels ----------------
__global__ void k_f32_to_f16(const float* __restrict__ s, _Float16* __restrict__ d, int n) {
    int i = blockIdx.x * 256 + threadIdx.x;
    if (i < n) d[i] = (_Float16)s[i];
}

// dt_w [1536,48] f32 -> [1536,64] f16 zero-padded
__global__ void k_pad_dtw(const float* __restrict__ s, _Float16* __restrict__ d) {
    int i = blockIdx.x * 256 + threadIdx.x;
    if (i >= D_INNER * DT_PAD) return;
    int r = i / DT_PAD, c = i % DT_PAD;
    d[i] = (c < DT_RANK) ? (_Float16)s[r * DT_RANK + c] : (_Float16)0.0f;
}

// xproj_w [80,1536] f32 -> [128,1536] f16 zero-padded rows
__global__ void k_pad_xproj(const float* __restrict__ s, _Float16* __restrict__ d) {
    int i = blockIdx.x * 256 + threadIdx.x;
    if (i >= XPROJ_NP * D_INNER) return;
    int r = i / D_INNER;
    d[i] = (r < XPROJ_N) ? (_Float16)s[i] : (_Float16)0.0f;
}

// dbc[:, :48] f32 (ld=128) -> dt48 [*,64] f16 zero-padded (real rows)
__global__ void k_pack_dt(const float* __restrict__ dbc, _Float16* __restrict__ d) {
    int i = blockIdx.x * 256 + threadIdx.x;
    if (i >= MROWS * DT_PAD) return;
    int m = i / DT_PAD, c = i % DT_PAD;
    d[(size_t)m * DT_PAD + c] = (c < DT_RANK) ? (_Float16)dbc[(size_t)m * DBC_LD + c]
                                              : (_Float16)0.0f;
}

__global__ void k_aneg(const float* __restrict__ Alog, float* __restrict__ An) {
    int i = blockIdx.x * 256 + threadIdx.x;
    if (i < D_INNER * D_STATE) An[i] = -__expf(Alog[i]);
}

// ---------------- timestep embedding (tiny; VALU) ----------------
__global__ void k_temb_embed(const int* __restrict__ t, float* __restrict__ e) {
    int i = blockIdx.x * 256 + threadIdx.x;
    if (i >= BATCH * D_MODEL) return;
    int b = i / D_MODEL, d = i % D_MODEL;
    const int half = D_MODEL / 2;
    int j = (d < half) ? d : d - half;
    float f = __expf((-9.210340371976184f / (float)(half - 1)) * (float)j);
    float a = (float)t[b] * f;
    e[i] = (d < half) ? sinf(a) : cosf(a);
}
__global__ void k_temb_h(const float* __restrict__ e, const float* __restrict__ w1,
                         const float* __restrict__ b1, float* __restrict__ th) {
    int i = blockIdx.x * 256 + threadIdx.x;
    if (i >= BATCH * 4 * D_MODEL) return;
    int b = i / (4 * D_MODEL), j = i % (4 * D_MODEL);
    const float* er = e + (size_t)b * D_MODEL;
    const float* wr = w1 + (size_t)j * D_MODEL;
    float a = b1[j];
    for (int d = 0; d < D_MODEL; ++d) a += er[d] * wr[d];
    th[i] = siluf(a);
}
__global__ void k_temb_o(const float* __restrict__ th, const float* __restrict__ w2,
                         const float* __restrict__ b2, float* __restrict__ tv) {
    int i = blockIdx.x * 256 + threadIdx.x;
    if (i >= BATCH * D_MODEL) return;
    int b = i / D_MODEL, d = i % D_MODEL;
    const float* hr = th + (size_t)b * 4 * D_MODEL;
    const float* wr = w2 + (size_t)d * 4 * D_MODEL;
    float a = b2[d];
    for (int j = 0; j < 4 * D_MODEL; ++j) a += hr[j] * wr[j];
    tv[i] = a;
}
__global__ void k_add_temb(const float* __restrict__ x, const float* __restrict__ tv,
                           float* __restrict__ xo) {
    int i = blockIdx.x * 256 + threadIdx.x;
    if (i >= MROWS * D_MODEL) return;
    int m = i / D_MODEL, d = i % D_MODEL;
    xo[i] = x[i] + tv[(m / SEQL) * D_MODEL + d];
}

// ---------------- LayerNorm (768 cols, block per row, f16 out) ----------------
__global__ __launch_bounds__(256)
void k_ln_f16(const float* __restrict__ x, const float* __restrict__ g,
              const float* __restrict__ bt, _Float16* __restrict__ o) {
    const int row = blockIdx.x;
    const int tid = threadIdx.x;
    const float* xr = x + (size_t)row * D_MODEL;
    float v0 = xr[tid], v1 = xr[tid + 256], v2 = xr[tid + 512];
    float s = v0 + v1 + v2;
#pragma unroll
    for (int off = 16; off; off >>= 1) s += __shfl_xor(s, off, 32);
    __shared__ float red[8];
    int wv = tid >> 5, ln = tid & 31;
    if (ln == 0) red[wv] = s;
    __syncthreads();
    float mu = 0.f;
#pragma unroll
    for (int w = 0; w < 8; ++w) mu += red[w];
    mu *= (1.0f / (float)D_MODEL);
    float d0 = v0 - mu, d1 = v1 - mu, d2 = v2 - mu;
    float q = d0 * d0 + d1 * d1 + d2 * d2;
#pragma unroll
    for (int off = 16; off; off >>= 1) q += __shfl_xor(q, off, 32);
    __syncthreads();
    if (ln == 0) red[wv] = q;
    __syncthreads();
    float var = 0.f;
#pragma unroll
    for (int w = 0; w < 8; ++w) var += red[w];
    var *= (1.0f / (float)D_MODEL);
    float rs = rsqrtf(var + 1e-5f);
    _Float16* orow = o + (size_t)row * D_MODEL;
    orow[tid]       = (_Float16)(d0 * rs * g[tid]       + bt[tid]);
    orow[tid + 256] = (_Float16)(d1 * rs * g[tid + 256] + bt[tid + 256]);
    orow[tid + 512] = (_Float16)(d2 * rs * g[tid + 512] + bt[tid + 512]);
}

// ---------------- causal depthwise conv + SiLU (dir handles flipped mamba) ----
__global__ void k_conv_silu(const float* __restrict__ xz, const float* __restrict__ cw,
                            const float* __restrict__ cb, int dir,
                            float* __restrict__ xc, _Float16* __restrict__ xch) {
    int i = blockIdx.x * 256 + threadIdx.x;
    if (i >= MROWS * D_INNER) return;
    int m = i / D_INNER, d = i % D_INNER;
    int b = m / SEQL, s = m % SEQL;
    float acc = cb[d];
#pragma unroll
    for (int k = 0; k < D_CONV; ++k) {
        int ss = dir ? (s + 3 - k) : (s - 3 + k);
        if (ss >= 0 && ss < SEQL)
            acc += cw[d * D_CONV + k] * xz[(size_t)(b * SEQL + ss) * (2 * D_INNER) + d];
    }
    float v = siluf(acc);
    xc[i] = v;
    xch[i] = (_Float16)v;
}

// ---------------- selective scan (sequential over S; 1 thread / (b,d)) -------
__global__ __launch_bounds__(256)
void k_scan(const float* __restrict__ dt, const float* __restrict__ dbc,
            const float* __restrict__ xc, const float* __restrict__ xz,
            const float* __restrict__ An, const float* __restrict__ Dp,
            int dir, _Float16* __restrict__ y16) {
    int t = blockIdx.x * 256 + threadIdx.x;
    if (t >= BATCH * D_INNER) return;
    int b = t / D_INNER, d = t % D_INNER;
    float a[D_STATE], h[D_STATE];
#pragma unroll
    for (int n = 0; n < D_STATE; ++n) { a[n] = An[d * D_STATE + n]; h[n] = 0.f; }
    const float Dd = Dp[d];
    for (int i = 0; i < SEQL; ++i) {
        int s = dir ? (SEQL - 1 - i) : i;
        size_t m = (size_t)(b * SEQL + s);
        float dtv = dt[m * D_INNER + d];
        float xv  = xc[m * D_INNER + d];
        float zv  = xz[m * (2 * D_INNER) + D_INNER + d];
        float du  = dtv * xv;
        const float* bc = dbc + m * DBC_LD;
        float acc = 0.f;
#pragma unroll
        for (int n = 0; n < D_STATE; ++n) {
            h[n] = __expf(dtv * a[n]) * h[n] + du * bc[DT_RANK + n];
            acc += h[n] * bc[DT_RANK + D_STATE + n];
        }
        float y = (acc + Dd * xv) * siluf(zv);
        y16[m * D_INNER + d] = (_Float16)y;
    }
}

// ---------------- cross-attention: one wave per (row, head) ------------------
__global__ __launch_bounds__(256)
void k_attn(const _Float16* __restrict__ q16, const float* __restrict__ kv,
            _Float16* __restrict__ o16) {
    const int m = blockIdx.x;           // 0..MROWS-1
    const int h = threadIdx.x >> 5;     // 8 heads = 8 waves
    const int ln = threadIdx.x & 31;
    const int b = m / SEQL;
    const _Float16* qr = q16 + (size_t)m * D_MODEL + h * DH;
    const float q0 = (float)qr[ln], q1 = (float)qr[ln + 32], q2 = (float)qr[ln + 64];
    const float* Kb = kv + (size_t)(b * COND_LEN) * (2 * D_MODEL) + h * DH;
    const float* Vb = Kb + D_MODEL;
    const float scale = 0.10206207261596577f; // 1/sqrt(96)
    float mx = -3.0e38f, sm = 0.f, o0 = 0.f, o1 = 0.f, o2 = 0.f;
    for (int c = 0; c < COND_LEN; ++c) {
        const float* kr = Kb + (size_t)c * (2 * D_MODEL);
        float p = q0 * kr[ln] + q1 * kr[ln + 32] + q2 * kr[ln + 64];
#pragma unroll
        for (int off = 16; off; off >>= 1) p += __shfl_xor(p, off, 32);
        float s = p * scale;
        float nmx = fmaxf(mx, s);
        float corr = __expf(mx - nmx);
        float w = __expf(s - nmx);
        const float* vr = Vb + (size_t)c * (2 * D_MODEL);
        sm = sm * corr + w;
        o0 = o0 * corr + w * vr[ln];
        o1 = o1 * corr + w * vr[ln + 32];
        o2 = o2 * corr + w * vr[ln + 64];
        mx = nmx;
    }
    float inv = 1.0f / sm;
    _Float16* orow = o16 + (size_t)m * D_MODEL + h * DH;
    orow[ln]      = (_Float16)(o0 * inv);
    orow[ln + 32] = (_Float16)(o1 * inv);
    orow[ln + 64] = (_Float16)(o2 * inv);
}

// ---------------- head: [M,768] f16 -> [M,9] f32 -----------------------------
__global__ void k_head(const _Float16* __restrict__ xn, const float* __restrict__ hw,
                       const float* __restrict__ hb, float* __restrict__ out) {
    int i = blockIdx.x * 256 + threadIdx.x;
    if (i >= MROWS * EDGE_DIM) return;
    int m = i / EDGE_DIM, e = i % EDGE_DIM;
    const _Float16* xr = xn + (size_t)m * D_MODEL;
    const float* wr = hw + (size_t)e * D_MODEL;
    float acc = hb[e];
    for (int d = 0; d < D_MODEL; ++d) acc += (float)xr[d] * wr[d];
    out[i] = acc;
}

// =================================================================
// host side
// =================================================================
static void gemm(int epi, hipStream_t st, const _Float16* A, int lda,
                 const _Float16* W, int ldw, int M, int N, int K,
                 const float* bias, const float* R, int ldr,
                 float* Cf, int ldc, _Float16* Ch, int ldch)
{
    dim3 grid(CDIV(N, 256), CDIV(M, 32)), block(256);
    switch (epi) {
    case 0: gemm_wmma_f16<0><<<grid, block, 0, st>>>(A,lda,W,ldw,M,N,K,bias,R,ldr,Cf,ldc,Ch,ldch); break;
    case 1: gemm_wmma_f16<1><<<grid, block, 0, st>>>(A,lda,W,ldw,M,N,K,bias,R,ldr,Cf,ldc,Ch,ldch); break;
    case 2: gemm_wmma_f16<2><<<grid, block, 0, st>>>(A,lda,W,ldw,M,N,K,bias,R,ldr,Cf,ldc,Ch,ldch); break;
    case 3: gemm_wmma_f16<3><<<grid, block, 0, st>>>(A,lda,W,ldw,M,N,K,bias,R,ldr,Cf,ldc,Ch,ldch); break;
    case 4: gemm_wmma_f16<4><<<grid, block, 0, st>>>(A,lda,W,ldw,M,N,K,bias,R,ldr,Cf,ldc,Ch,ldch); break;
    case 5: gemm_wmma_f16<5><<<grid, block, 0, st>>>(A,lda,W,ldw,M,N,K,bias,R,ldr,Cf,ldc,Ch,ldch); break;
    case 6: gemm_wmma_f16<6><<<grid, block, 0, st>>>(A,lda,W,ldw,M,N,K,bias,R,ldr,Cf,ldc,Ch,ldch); break;
    }
}

struct MambaRaw { const float *in_w,*conv_w,*conv_b,*xproj_w,*dt_w,*dt_b,*A_log,*D,*out_w; };
struct Mamba16  { _Float16 *in16,*xproj16,*dtw16,*out16; };

extern "C" void kernel_launch(void* const* d_in, const int* in_sizes, int n_in,
                              void* d_out, int out_size, void* d_ws, size_t ws_size,
                              hipStream_t stream)
{
    (void)in_sizes; (void)n_in; (void)out_size; (void)ws_size;
    int pi = 0;
    auto nextp = [&]() { return (const float*)d_in[pi++]; };

    const float* x_in  = nextp();
    const float* cond  = nextp();
    const int*   tstep = (const int*)d_in[pi++];

    const float *temb_w1 = nextp(), *temb_b1 = nextp(), *temb_w2 = nextp(), *temb_b2 = nextp();

    struct LayerRaw {
        const float *m_ng,*m_nb; MambaRaw fwd,bwd;
        const float *mix_w,*mix_b,*fn_g,*fn_b,*mf_w1,*mf_b1,*mf_w2,*mf_b2;
        const float *c_ng,*c_nb,*c_inw,*c_inb,*c_ow,*c_ob;
        const float *f_ng,*f_nb,*f_w1,*f_b1,*f_w2,*f_b2;
    } L[N_LAYERS];
    auto mraw = [&]() {
        MambaRaw m;
        m.in_w=nextp(); m.conv_w=nextp(); m.conv_b=nextp(); m.xproj_w=nextp();
        m.dt_w=nextp(); m.dt_b=nextp(); m.A_log=nextp(); m.D=nextp(); m.out_w=nextp();
        return m;
    };
    for (int l = 0; l < N_LAYERS; ++l) {
        L[l].m_ng=nextp(); L[l].m_nb=nextp(); L[l].fwd=mraw(); L[l].bwd=mraw();
        L[l].mix_w=nextp(); L[l].mix_b=nextp(); L[l].fn_g=nextp(); L[l].fn_b=nextp();
        L[l].mf_w1=nextp(); L[l].mf_b1=nextp(); L[l].mf_w2=nextp(); L[l].mf_b2=nextp();
        L[l].c_ng=nextp(); L[l].c_nb=nextp(); L[l].c_inw=nextp(); L[l].c_inb=nextp();
        L[l].c_ow=nextp(); L[l].c_ob=nextp();
        L[l].f_ng=nextp(); L[l].f_nb=nextp();
        L[l].f_w1=nextp(); L[l].f_b1=nextp(); L[l].f_w2=nextp(); L[l].f_b2=nextp();
    }
    const float *fin_g = nextp(), *fin_b = nextp(), *head_w = nextp(), *head_b = nextp();

    // ---- bump allocator over workspace ----
    char* wsb = (char*)d_ws;
    size_t off = 0;
    auto alloc = [&](size_t bytes) -> void* {
        off = (off + 255) & ~(size_t)255;
        void* p = wsb + off;
        off += bytes;
        return p;
    };

    // activations
    float*    xbuf  = (float*)   alloc((size_t)MP * D_MODEL * 4);
    _Float16* xn16  = (_Float16*)alloc((size_t)MP * D_MODEL * 2);
    float*    xz    = (float*)   alloc((size_t)MP * 2 * D_INNER * 4);
    float*    xc    = (float*)   alloc((size_t)MP * D_INNER * 4);
    _Float16* xc16  = (_Float16*)alloc((size_t)MP * D_INNER * 2);
    float*    dbc   = (float*)   alloc((size_t)MP * DBC_LD * 4);
    _Float16* dt48  = (_Float16*)alloc((size_t)MP * DT_PAD * 2);
    float*    dtb   = (float*)   alloc((size_t)MP * D_INNER * 4);
    _Float16* y16   = (_Float16*)alloc((size_t)MP * D_INNER * 2);
    _Float16* fb16  = (_Float16*)alloc((size_t)MP * D_INNER * 2);
    _Float16* h16   = (_Float16*)alloc((size_t)MP * 4 * D_MODEL * 2);
    _Float16* q16   = (_Float16*)alloc((size_t)MP * D_MODEL * 2);
    _Float16* o16   = (_Float16*)alloc((size_t)MP * D_MODEL * 2);
    float*    kv    = (float*)   alloc((size_t)MC * 2 * D_MODEL * 4);
    _Float16* cond16= (_Float16*)alloc((size_t)MC * D_MODEL * 2);
    float*    Aneg  = (float*)   alloc((size_t)D_INNER * D_STATE * 4);
    float*    tE    = (float*)   alloc((size_t)BATCH * D_MODEL * 4);
    float*    tH    = (float*)   alloc((size_t)BATCH * 4 * D_MODEL * 4);
    float*    tV    = (float*)   alloc((size_t)BATCH * D_MODEL * 4);

    // ---- convert weights to f16 ----
    auto cvt = [&](const float* src, size_t n) -> _Float16* {
        _Float16* dst = (_Float16*)alloc(n * 2);
        k_f32_to_f16<<<CDIV((int)n, 256), 256, 0, stream>>>(src, dst, (int)n);
        return dst;
    };
    auto m16 = [&](const MambaRaw& m) {
        Mamba16 w;
        w.in16    = cvt(m.in_w, (size_t)2 * D_INNER * D_MODEL);
        w.xproj16 = (_Float16*)alloc((size_t)XPROJ_NP * D_INNER * 2);
        k_pad_xproj<<<CDIV(XPROJ_NP * D_INNER, 256), 256, 0, stream>>>(m.xproj_w, w.xproj16);
        w.dtw16   = (_Float16*)alloc((size_t)D_INNER * DT_PAD * 2);
        k_pad_dtw<<<CDIV(D_INNER * DT_PAD, 256), 256, 0, stream>>>(m.dt_w, w.dtw16);
        w.out16   = cvt(m.out_w, (size_t)D_MODEL * D_INNER);
        return w;
    };
    struct Layer16 {
        Mamba16 fwd, bwd;
        _Float16 *mix16,*mfw1,*mfw2,*cin16,*cout16,*fw1,*fw2;
    } W16[N_LAYERS];
    for (int l = 0; l < N_LAYERS; ++l) {
        W16[l].fwd   = m16(L[l].fwd);
        W16[l].bwd   = m16(L[l].bwd);
        W16[l].mix16 = cvt(L[l].mix_w, (size_t)D_MODEL * D_INNER);
        W16[l].mfw1  = cvt(L[l].mf_w1, (size_t)4 * D_MODEL * D_MODEL);
        W16[l].mfw2  = cvt(L[l].mf_w2, (size_t)D_MODEL * 4 * D_MODEL);
        W16[l].cin16 = cvt(L[l].c_inw, (size_t)3 * D_MODEL * D_MODEL);
        W16[l].cout16= cvt(L[l].c_ow,  (size_t)D_MODEL * D_MODEL);
        W16[l].fw1   = cvt(L[l].f_w1,  (size_t)4 * D_MODEL * D_MODEL);
        W16[l].fw2   = cvt(L[l].f_w2,  (size_t)D_MODEL * 4 * D_MODEL);
    }
    // conditioning -> f16 (used by every layer's K/V projection)
    k_f32_to_f16<<<CDIV(MC * D_MODEL, 256), 256, 0, stream>>>(cond, cond16, MC * D_MODEL);

    // ---- timestep embedding + add ----
    k_temb_embed<<<CDIV(BATCH * D_MODEL, 256), 256, 0, stream>>>(tstep, tE);
    k_temb_h<<<CDIV(BATCH * 4 * D_MODEL, 256), 256, 0, stream>>>(tE, temb_w1, temb_b1, tH);
    k_temb_o<<<CDIV(BATCH * D_MODEL, 256), 256, 0, stream>>>(tH, temb_w2, temb_b2, tV);
    k_add_temb<<<CDIV(MROWS * D_MODEL, 256), 256, 0, stream>>>(x_in, tV, xbuf);

    // ---- per-direction mamba ----
    auto run_mamba = [&](const MambaRaw& r, const Mamba16& w, int dir, _Float16* fb_dst) {
        // xz = xn @ in_w^T           [MP, 3072]
        gemm(0, stream, xn16, D_MODEL, w.in16, D_MODEL, MP, 2 * D_INNER, D_MODEL,
             nullptr, nullptr, 0, xz, 2 * D_INNER, nullptr, 0);
        // xc = silu(causal dwconv(xi))
        k_conv_silu<<<CDIV(MROWS * D_INNER, 256), 256, 0, stream>>>(xz, r.conv_w, r.conv_b, dir, xc, xc16);
        // dbc = xc @ xproj^T         [MP, 128] (cols >= 80 are zero-weight pad)
        gemm(0, stream, xc16, D_INNER, w.xproj16, D_INNER, MP, XPROJ_NP, D_INNER,
             nullptr, nullptr, 0, dbc, DBC_LD, nullptr, 0);
        k_pack_dt<<<CDIV(MROWS * DT_PAD, 256), 256, 0, stream>>>(dbc, dt48);
        // dt = softplus(dt_raw @ dt_w^T + dt_b)   [MP, 1536]
        gemm(4, stream, dt48, DT_PAD, w.dtw16, DT_PAD, MP, D_INNER, DT_PAD,
             r.dt_b, nullptr, 0, dtb, D_INNER, nullptr, 0);
        k_aneg<<<CDIV(D_INNER * D_STATE, 256), 256, 0, stream>>>(r.A_log, Aneg);
        k_scan<<<CDIV(BATCH * D_INNER, 256), 256, 0, stream>>>(dtb, dbc, xc, xz, Aneg, r.D, dir, y16);
        // f/bw = y @ out_w^T -> f16 into concat buffer (no flip needed: row-wise)
        gemm(5, stream, y16, D_INNER, w.out16, D_INNER, MP, D_MODEL, D_INNER,
             nullptr, nullptr, 0, nullptr, 0, fb_dst, D_INNER);
    };

    for (int l = 0; l < N_LAYERS; ++l) {
        // ===== mamba block =====
        k_ln_f16<<<MROWS, 256, 0, stream>>>(xbuf, L[l].m_ng, L[l].m_nb, xn16);
        run_mamba(L[l].fwd, W16[l].fwd, 0, fb16);
        run_mamba(L[l].bwd, W16[l].bwd, 1, fb16 + D_MODEL);
        // x = concat(f,bw) @ mix^T + mix_b + x
        gemm(3, stream, fb16, D_INNER, W16[l].mix16, D_INNER, MP, D_MODEL, D_INNER,
             L[l].mix_b, xbuf, D_MODEL, xbuf, D_MODEL, nullptr, 0);
        // x += ffn(ln(x))
        k_ln_f16<<<MROWS, 256, 0, stream>>>(xbuf, L[l].fn_g, L[l].fn_b, xn16);
        gemm(2, stream, xn16, D_MODEL, W16[l].mfw1, D_MODEL, MP, 4 * D_MODEL, D_MODEL,
             L[l].mf_b1, nullptr, 0, nullptr, 0, h16, 4 * D_MODEL);
        gemm(3, stream, h16, 4 * D_MODEL, W16[l].mfw2, 4 * D_MODEL, MP, D_MODEL, 4 * D_MODEL,
             L[l].mf_b2, xbuf, D_MODEL, xbuf, D_MODEL, nullptr, 0);
        // ===== cross attention =====
        k_ln_f16<<<MROWS, 256, 0, stream>>>(xbuf, L[l].c_ng, L[l].c_nb, xn16);
        gemm(6, stream, xn16, D_MODEL, W16[l].cin16, D_MODEL, MP, D_MODEL, D_MODEL,
             L[l].c_inb, nullptr, 0, nullptr, 0, q16, D_MODEL);                        // Q
        gemm(1, stream, cond16, D_MODEL, W16[l].cin16 + (size_t)D_MODEL * D_MODEL, D_MODEL,
             MC, 2 * D_MODEL, D_MODEL, L[l].c_inb + D_MODEL, nullptr, 0,
             kv, 2 * D_MODEL, nullptr, 0);                                             // K,V
        k_attn<<<MROWS, 256, 0, stream>>>(q16, kv, o16);
        gemm(3, stream, o16, D_MODEL, W16[l].cout16, D_MODEL, MP, D_MODEL, D_MODEL,
             L[l].c_ob, xbuf, D_MODEL, xbuf, D_MODEL, nullptr, 0);
        // ===== ffn =====
        k_ln_f16<<<MROWS, 256, 0, stream>>>(xbuf, L[l].f_ng, L[l].f_nb, xn16);
        gemm(2, stream, xn16, D_MODEL, W16[l].fw1, D_MODEL, MP, 4 * D_MODEL, D_MODEL,
             L[l].f_b1, nullptr, 0, nullptr, 0, h16, 4 * D_MODEL);
        gemm(3, stream, h16, 4 * D_MODEL, W16[l].fw2, 4 * D_MODEL, MP, D_MODEL, 4 * D_MODEL,
             L[l].f_b2, xbuf, D_MODEL, xbuf, D_MODEL, nullptr, 0);
    }

    // ---- final norm + head ----
    k_ln_f16<<<MROWS, 256, 0, stream>>>(xbuf, fin_g, fin_b, xn16);
    k_head<<<CDIV(MROWS * EDGE_DIM, 256), 256, 0, stream>>>(xn16, head_w, head_b, (float*)d_out);
}
